// NeuralCDE_88003879895524
// MI455X (gfx1250) — compile-verified, matrix operand order checked by reading, not run on hardware
//
#include <hip/hip_runtime.h>
#include <cmath>

typedef __attribute__((ext_vector_type(16))) __bf16 v16bf;
typedef __attribute__((ext_vector_type(8)))  float  v8f;

#define NWAVES 8
#define ACT_STRIDE 132   // padded row stride (floats) to dodge LDS bank conflicts

// workspace offsets in bf16 elements (swizzled weight fragments)
constexpr size_t OFF_VF_W0   = 0;        // NT=8,  KT=2 ->   8192
constexpr size_t OFF_VF_WH   = 8192;     // NT=8,  KT=4 ->  16384
constexpr size_t OFF_VF_WOUT = 24576;    // NT=64, KT=4 -> 131072
constexpr size_t OFF_IE_W0   = 155648;   // NT=8,  KT=2 ->   8192
constexpr size_t OFF_IE_WH   = 163840;   // NT=8,  KT=4 ->  16384
constexpr size_t OFF_IE_WOUT = 180224;   // NT=4,  KT=4 ->   8192

__constant__ float A_TAB[6][5] = {
  {0.f,0.f,0.f,0.f,0.f},
  {0.2f,0.f,0.f,0.f,0.f},
  {3.f/40.f, 9.f/40.f, 0.f,0.f,0.f},
  {44.f/45.f, -56.f/15.f, 32.f/9.f, 0.f,0.f},
  {19372.f/6561.f, -25360.f/2187.f, 64448.f/6561.f, -212.f/729.f, 0.f},
  {9017.f/3168.f, -355.f/33.f, 46732.f/5247.f, 49.f/176.f, -5103.f/18656.f}
};
__constant__ float C_TAB[6] = {0.f, 0.2f, 0.3f, 0.8f, 8.f/9.f, 1.f};
__constant__ float B_TAB[6] = {35.f/384.f, 0.f, 500.f/1113.f, 125.f/192.f,
                               -2187.f/6784.f, 11.f/84.f};

// ---------------------------------------------------------------------------
// Weight swizzle: f32 W[O][Kin] -> bf16 fragments laid out exactly as the
// 16-bit 32x16 B operand expects per-lane (ISA 7.12.2), zero-padding K.
// dst[((kt*NT + nt)*32 + lane)*16 + e]
// ---------------------------------------------------------------------------
__global__ void swizzle_w(const float* __restrict__ W, __bf16* __restrict__ dst,
                          int Kin, int KT, int NT) {
  int idx = blockIdx.x * 256 + threadIdx.x;
  int total = NT * KT * 512;
  if (idx >= total) return;
  int e     = idx & 15;
  int lane  = (idx >> 4) & 31;
  int tile  = idx >> 9;
  int nt    = tile % NT;
  int kt    = tile / NT;
  int n     = lane & 15;
  int khalf = lane >> 4;
  int k  = (e < 8) ? (8*khalf + e) : (16 + 8*khalf + (e - 8));
  int kg = kt*32 + k;
  int ng = nt*16 + n;
  float v = (kg < Kin) ? W[(size_t)ng * Kin + kg] : 0.0f;
  dst[idx] = (__bf16)v;
}

// A fragment: lane holds row m = lane%16; f32 activations gathered from LDS
// and converted to bf16 per the 16-bit 16x32 A layout.
__device__ inline v16bf load_afrag(const float* act, int lane, int kt) {
  int m = lane & 15;
  int khalf = lane >> 4;
  const float* row = act + m*ACT_STRIDE + kt*32 + 8*khalf;
  v16bf a;
#pragma unroll
  for (int e = 0; e < 8; ++e) a[e]   = (__bf16)row[e];
#pragma unroll
  for (int e = 0; e < 8; ++e) a[8+e] = (__bf16)row[16 + e];
  return a;
}

// Batched 16xK * KxN GEMM via WMMA; bias + activation; writes f32 tile to LDS.
// actfn: 0=relu, 1=softplus, 2=identity
template<int KT, int NT>
__device__ inline void wmma_layer(const float* __restrict__ actIn,
                                  const __bf16* __restrict__ wsz,
                                  const float* __restrict__ bias,
                                  float* __restrict__ actOut,
                                  int actfn, int lane, int wave) {
  v16bf afr[KT];
#pragma unroll
  for (int kt = 0; kt < KT; ++kt) afr[kt] = load_afrag(actIn, lane, kt);
  int n = lane & 15, khalf = lane >> 4;
  for (int nt = wave; nt < NT; nt += NWAVES) {
    v8f acc = {0.f,0.f,0.f,0.f,0.f,0.f,0.f,0.f};
#pragma unroll
    for (int kt = 0; kt < KT; ++kt) {
      v16bf bfr = *(const v16bf*)(wsz + (((size_t)(kt*NT + nt)) << 9) + (lane << 4));
      acc = __builtin_amdgcn_wmma_f32_16x16x32_bf16(false, afr[kt], false, bfr,
                                                    (short)0, acc, false, false);
    }
    float bv = bias[nt*16 + n];
#pragma unroll
    for (int r = 0; r < 8; ++r) {
      float v = acc[r] + bv;
      if (actfn == 0)      v = fmaxf(v, 0.f);
      else if (actfn == 1) v = (v > 20.f) ? v : log1pf(expf(v));
      actOut[(r + 8*khalf)*ACT_STRIDE + nt*16 + n] = v;
    }
  }
}

// Fused vf output layer: [16x128]x[128x1024] tiles; N-tile nt == latent l,
// columns are channels c. tanh, scale by dx[sample][c], reduce over the 16
// N-lanes via shuffles straight into k-stage buffer kout[sample*64 + l].
__device__ inline void wmma_vfout(const float* __restrict__ actIn,
                                  const __bf16* __restrict__ wsz,
                                  const float* __restrict__ bias,
                                  const float* __restrict__ dxbuf,
                                  float* __restrict__ kout,
                                  int lane, int wave) {
  v16bf afr[4];
#pragma unroll
  for (int kt = 0; kt < 4; ++kt) afr[kt] = load_afrag(actIn, lane, kt);
  int c = lane & 15, khalf = lane >> 4;
  for (int nt = wave; nt < 64; nt += NWAVES) {
    v8f acc = {0.f,0.f,0.f,0.f,0.f,0.f,0.f,0.f};
#pragma unroll
    for (int kt = 0; kt < 4; ++kt) {
      v16bf bfr = *(const v16bf*)(wsz + (((size_t)(kt*64 + nt)) << 9) + (lane << 4));
      acc = __builtin_amdgcn_wmma_f32_16x16x32_bf16(false, afr[kt], false, bfr,
                                                    (short)0, acc, false, false);
    }
    float bv = bias[nt*16 + c];
#pragma unroll
    for (int r = 0; r < 8; ++r) {
      int m = r + 8*khalf;
      float v = tanhf(acc[r] + bv) * dxbuf[m*16 + c];
#pragma unroll
      for (int off = 1; off < 16; off <<= 1) v += __shfl_xor(v, off, 32);
      if (c == 0) kout[m*64 + nt] = v;
    }
  }
}

__device__ inline void decode_step(const float* __restrict__ sYv,
                                   float* __restrict__ h1, float* __restrict__ h2,
                                   const float* __restrict__ dw0, const float* __restrict__ db0,
                                   const float* __restrict__ dwh, const float* __restrict__ dbh,
                                   const float* __restrict__ dwo, const float* __restrict__ dbo,
                                   float* __restrict__ out, int b0, int tstep, int tid) {
  for (int p = tid; p < 512; p += 256) {
    int s = p >> 5, u = p & 31;
    const float* wr = dw0 + u*64;
    const float* yr = sYv + s*64;
    float a = db0[u];
#pragma unroll 8
    for (int k = 0; k < 64; ++k) a += wr[k]*yr[k];
    h1[p] = fmaxf(a, 0.f);
  }
  __syncthreads();
  for (int p = tid; p < 512; p += 256) {
    int s = p >> 5, u = p & 31;
    const float* wr = dwh + u*32;
    const float* hr = h1 + s*32;
    float a = dbh[u];
#pragma unroll
    for (int k = 0; k < 32; ++k) a += wr[k]*hr[k];
    h2[p] = fmaxf(a, 0.f);
  }
  __syncthreads();
  if (tid < 128) {
    int s = tid >> 3, o = tid & 7;
    const float* wr = dwo + o*32;
    const float* hr = h2 + s*32;
    float a = dbo[o];
#pragma unroll
    for (int k = 0; k < 32; ++k) a += wr[k]*hr[k];
    out[(((size_t)(b0+s))*128 + tstep)*8 + o] = a;
  }
  __syncthreads();
}

// ---------------------------------------------------------------------------
// Persistent fused kernel: one block integrates 16 samples end-to-end.
// ---------------------------------------------------------------------------
__global__ __launch_bounds__(256, 1)
void ncde_main(const float* __restrict__ ts,      const float* __restrict__ obs,
               const float* __restrict__ coefD,   const float* __restrict__ coefC,
               const float* __restrict__ coefB,   const float* __restrict__ coefA,
               const float* __restrict__ gru_wih, const float* __restrict__ gru_whh,
               const float* __restrict__ gru_bih, const float* __restrict__ gru_bhh,
               const float* __restrict__ gru_lin,
               const float* __restrict__ ie_b0,   const float* __restrict__ ie_bh,
               const float* __restrict__ ie_bout,
               const float* __restrict__ vf_b0,   const float* __restrict__ vf_bh,
               const float* __restrict__ vf_bout,
               const float* __restrict__ dec_w0,  const float* __restrict__ dec_b0,
               const float* __restrict__ dec_wh,  const float* __restrict__ dec_bh,
               const float* __restrict__ dec_wout,const float* __restrict__ dec_bout,
               const __bf16* __restrict__ ws,     float* __restrict__ out) {
  __shared__ float sY[1024];              // y state  [16][64]
  __shared__ float sK[6*1024];            // RK stages
  __shared__ float sAct0[16*ACT_STRIDE];  // activation ping
  __shared__ float sAct1[16*ACT_STRIDE];  // activation pong
  __shared__ float sDx[256];              // dx [16][16]
  __shared__ float sH[1024];              // GRU h
  __shared__ float sHn[1024];             // GRU h next

  const int tid  = threadIdx.x;
  const int lane = tid & 31;
  const int wave = tid >> 5;
  const int b0   = blockIdx.x * 16;

  // ---------------- GRU encoder (24 sequential steps, VALU) ----------------
  for (int i = tid; i < 1024; i += 256) sH[i] = 0.f;
  __syncthreads();
  for (int step = 0; step < 24; ++step) {
    for (int p = tid; p < 1024; p += 256) {
      int s = p >> 6, u = p & 63;
      const float* hrow = sH + s*64;
      float hr = gru_bhh[u], hz = gru_bhh[64+u], hn = gru_bhh[128+u];
      const float* w0r = gru_whh + (size_t)u*64;
      const float* w1r = gru_whh + (size_t)(64+u)*64;
      const float* w2r = gru_whh + (size_t)(128+u)*64;
#pragma unroll 8
      for (int k = 0; k < 64; ++k) {
        float hk = hrow[k];
        hr += w0r[k]*hk; hz += w1r[k]*hk; hn += w2r[k]*hk;
      }
      const float* xr = obs + ((size_t)(b0+s)*24 + step)*8;
      float ir = gru_bih[u], iz = gru_bih[64+u], inn = gru_bih[128+u];
#pragma unroll
      for (int k = 0; k < 8; ++k) {
        float xk = xr[k];
        ir  += gru_wih[u*8+k]*xk;
        iz  += gru_wih[(64+u)*8+k]*xk;
        inn += gru_wih[(128+u)*8+k]*xk;
      }
      float r  = 1.f/(1.f + expf(-(ir+hr)));
      float z  = 1.f/(1.f + expf(-(iz+hz)));
      float nn = tanhf(inn + r*hn);
      sHn[p] = (1.f - z)*nn + z*hrow[u];
    }
    __syncthreads();
    for (int p = tid; p < 1024; p += 256) sH[p] = sHn[p];
    __syncthreads();
  }

  // ---------------- enc + ie-MLP input (cols: enc 0..31, x0 32..47, 0 pad) --
  for (int i = tid; i < 16*ACT_STRIDE; i += 256) sAct0[i] = 0.f;
  __syncthreads();
  for (int p = tid; p < 512; p += 256) {
    int s = p >> 5, e = p & 31;
    const float* wr = gru_lin + e*64;
    const float* hrow = sH + s*64;
    float acc = 0.f;
#pragma unroll 8
    for (int k = 0; k < 64; ++k) acc += wr[k]*hrow[k];
    sAct0[s*ACT_STRIDE + e] = acc;
  }
  {
    int s = tid >> 4, c = tid & 15;
    sAct0[s*ACT_STRIDE + 32 + c] = coefA[((size_t)(b0+s)*127)*16 + c];
  }
  __syncthreads();

  // ---------------- initial encoder MLP (WMMA) ----------------
  wmma_layer<2,8>(sAct0, ws + OFF_IE_W0,  ie_b0,  sAct1, 0, lane, wave);
  __syncthreads();
  wmma_layer<4,8>(sAct1, ws + OFF_IE_WH,  ie_bh,  sAct0, 0, lane, wave);
  __syncthreads();
  wmma_layer<4,4>(sAct0, ws + OFF_IE_WOUT, ie_bout, sAct1, 2, lane, wave);
  __syncthreads();
  for (int i = tid; i < 1024; i += 256) {
    int s = i >> 6, l = i & 63;
    sY[i] = sAct1[s*ACT_STRIDE + l];
  }
  __syncthreads();

  decode_step(sY, sAct0, sAct1, dec_w0, dec_b0, dec_wh, dec_bh,
              dec_wout, dec_bout, out, b0, 0, tid);

  // ---------------- Dormand-Prince integration ----------------
  for (int t = 0; t < 127; ++t) {
    float t0 = ts[t], t1 = ts[t+1];
    float hstep = (t1 - t0) * 0.5f;
    for (int j = 0; j < 2; ++j) {
      float f0 = (float)j * hstep;
      for (int st = 0; st < 6; ++st) {
        // stage input yy = y + h * sum a[st][q] k_q  -> sAct0 cols 0..63
        for (int i = tid; i < 1024; i += 256) {
          int s = i >> 6, l = i & 63;
          float v = sY[i];
          for (int q = 0; q < st; ++q) v += hstep * A_TAB[st][q] * sK[q*1024 + i];
          sAct0[s*ACT_STRIDE + l] = v;
        }
        // dx at frac = f0 + c[st]*h  (one (sample,channel) per thread)
        {
          int s = tid >> 4, c = tid & 15;
          float frac = f0 + C_TAB[st]*hstep;
          size_t ci = ((size_t)(b0+s)*127 + t)*16 + c;
          sDx[tid] = coefB[ci] + 2.f*coefC[ci]*frac + 3.f*coefD[ci]*frac*frac;
        }
        __syncthreads();
        wmma_layer<2,8>(sAct0, ws + OFF_VF_W0, vf_b0, sAct1, 1, lane, wave);
        __syncthreads();
        wmma_layer<4,8>(sAct1, ws + OFF_VF_WH, vf_bh, sAct0, 1, lane, wave);
        __syncthreads();
        wmma_vfout(sAct0, ws + OFF_VF_WOUT, vf_bout, sDx, sK + st*1024, lane, wave);
        __syncthreads();
      }
      // y += h * (b . k)
      for (int i = tid; i < 1024; i += 256) {
        float v = sY[i];
#pragma unroll
        for (int q = 0; q < 6; ++q) v += hstep * B_TAB[q] * sK[q*1024 + i];
        sY[i] = v;
      }
      __syncthreads();
    }
    decode_step(sY, sAct0, sAct1, dec_w0, dec_b0, dec_wh, dec_bh,
                dec_wout, dec_bout, out, b0, t + 1, tid);
  }
}

// ---------------------------------------------------------------------------
extern "C" void kernel_launch(void* const* d_in, const int* in_sizes, int n_in,
                              void* d_out, int out_size, void* d_ws, size_t ws_size,
                              hipStream_t stream) {
  const float* ts       = (const float*)d_in[0];
  const float* obs      = (const float*)d_in[1];
  const float* coefD    = (const float*)d_in[2];
  const float* coefC    = (const float*)d_in[3];
  const float* coefB    = (const float*)d_in[4];
  const float* coefA    = (const float*)d_in[5];
  const float* gru_wih  = (const float*)d_in[6];
  const float* gru_whh  = (const float*)d_in[7];
  const float* gru_bih  = (const float*)d_in[8];
  const float* gru_bhh  = (const float*)d_in[9];
  const float* gru_lin  = (const float*)d_in[10];
  const float* ie_w0    = (const float*)d_in[11];
  const float* ie_b0    = (const float*)d_in[12];
  const float* ie_wh    = (const float*)d_in[13];
  const float* ie_bh    = (const float*)d_in[14];
  const float* ie_wout  = (const float*)d_in[15];
  const float* ie_bout  = (const float*)d_in[16];
  const float* vf_w0    = (const float*)d_in[17];
  const float* vf_b0    = (const float*)d_in[18];
  const float* vf_wh    = (const float*)d_in[19];
  const float* vf_bh    = (const float*)d_in[20];
  const float* vf_wout  = (const float*)d_in[21];
  const float* vf_bout  = (const float*)d_in[22];
  const float* dec_w0   = (const float*)d_in[23];
  const float* dec_b0   = (const float*)d_in[24];
  const float* dec_wh   = (const float*)d_in[25];
  const float* dec_bh   = (const float*)d_in[26];
  const float* dec_wout = (const float*)d_in[27];
  const float* dec_bout = (const float*)d_in[28];
  __bf16* ws = (__bf16*)d_ws;
  float*  out = (float*)d_out;

  auto swz = [&](const float* W, size_t off, int Kin, int KT, int NT) {
    int total = NT * KT * 512;
    swizzle_w<<<(total + 255)/256, 256, 0, stream>>>(W, ws + off, Kin, KT, NT);
  };
  swz(vf_w0,   OFF_VF_W0,   64, 2, 8);
  swz(vf_wh,   OFF_VF_WH,  128, 4, 8);
  swz(vf_wout, OFF_VF_WOUT,128, 4, 64);
  swz(ie_w0,   OFF_IE_W0,   48, 2, 8);
  swz(ie_wh,   OFF_IE_WH,  128, 4, 8);
  swz(ie_wout, OFF_IE_WOUT,128, 4, 4);

  ncde_main<<<16, 256, 0, stream>>>(ts, obs, coefD, coefC, coefB, coefA,
      gru_wih, gru_whh, gru_bih, gru_bhh, gru_lin,
      ie_b0, ie_bh, ie_bout, vf_b0, vf_bh, vf_bout,
      dec_w0, dec_b0, dec_wh, dec_bh, dec_wout, dec_bout, ws, out);
}
